// EfficientViTAttention_24876450579163
// MI455X (gfx1250) — matrix operand order repaired and checked by reading, not compiled, not run on hardware
//
#include <hip/hip_runtime.h>
#include <hip/hip_bf16.h>
#include <math.h>

typedef __attribute__((ext_vector_type(16))) _Float16 v16h;
typedef __attribute__((ext_vector_type(8)))  _Float16 half8;
typedef __attribute__((ext_vector_type(4)))  _Float16 half4;
typedef __attribute__((ext_vector_type(8)))  float    v8f;
typedef __attribute__((ext_vector_type(4)))  float    f4v;

#define T_TOK   73728     // B*H*W = 8*96*96
#define LSEQ    9216      // H*W
#define CDIM    128
#define HEADS   8
#define DH      16

__device__ __forceinline__ float gelu_f(float x) {
    return 0.5f * x * (1.0f + erff(x * 0.70710678118654752f));
}

__device__ __forceinline__ v16h combine16(half8 a0, half8 a1) {
    return __builtin_shufflevector(a0, a1, 0, 1, 2, 3, 4, 5, 6, 7,
                                   8, 9, 10, 11, 12, 13, 14, 15);
}

// ---------------------------------------------------------------------------
// Weight repack: f32 [K][N] (optionally transposed source) -> WMMA B-fragment
// layout: dst[((kt*NT + nt)*32 + lane)*16 + j], k = kt*32 + (lane>>4)*8 + (j&7) + (j>>3)*16,
// n = nt*16 + (lane&15).  One contiguous 32B read per lane per fragment.
// ---------------------------------------------------------------------------
__global__ __launch_bounds__(256) void repack_w(const float* __restrict__ W,
                                                _Float16* __restrict__ dst,
                                                int K, int N, int transpose) {
    int idx = blockIdx.x * 256 + threadIdx.x;
    if (idx >= K * N) return;
    int j    = idx & 15;
    int lane = (idx >> 4) & 31;
    int rest = idx >> 9;
    int NT   = N >> 4;
    int nt   = rest % NT;
    int kt   = rest / NT;
    int k = kt * 32 + (lane >> 4) * 8 + (j & 7) + (j >> 3) * 16;
    int n = nt * 16 + (lane & 15);
    float v = transpose ? W[(size_t)n * K + k] : W[(size_t)k * N + n];
    dst[idx] = (_Float16)v;
}

// ---------------------------------------------------------------------------
// Depthwise 3x3 conv + bias + exact GELU, NCHW f32 in -> token-major f16 out.
// Block: (b, row h, 32-wide w-tile) x all 128 channels; LDS transpose so the
// token-major store is coalesced (two b128 f16 stores per thread).
// ---------------------------------------------------------------------------
__global__ __launch_bounds__(256) void dwconv_gelu(const float* __restrict__ x,
                                                   const float* __restrict__ dww,
                                                   const float* __restrict__ dwb,
                                                   _Float16* __restrict__ y) {
    __shared__ float sh[32 * CDIM];
    int bid = blockIdx.x;
    int wt = bid % 3;
    int h  = (bid / 3) % 96;
    int b  = bid / 288;
    int w0 = wt * 32;
    int wl = threadIdx.x & 31;
    int cg = threadIdx.x >> 5;      // 8 channel groups of 16
    int ww = w0 + wl;
    for (int cc = 0; cc < 16; ++cc) {
        int c = cg * 16 + cc;
        const float* xp = x + ((size_t)(b * CDIM + c) * 96) * 96;
        const float* wp = dww + c * 9;
        float acc = dwb[c];
        #pragma unroll
        for (int dh = -1; dh <= 1; ++dh) {
            int hh = h + dh;
            if (hh < 0 || hh >= 96) continue;
            #pragma unroll
            for (int dw = -1; dw <= 1; ++dw) {
                int w2 = ww + dw;
                if (w2 < 0 || w2 >= 96) continue;
                acc += wp[(dh + 1) * 3 + (dw + 1)] * xp[hh * 96 + w2];
            }
        }
        sh[wl * CDIM + c] = gelu_f(acc);
    }
    __syncthreads();
    int tw = threadIdx.x >> 3;            // token within tile
    int cs = (threadIdx.x & 7) * 16;      // channel start
    size_t t = (size_t)b * LSEQ + (size_t)h * 96 + w0 + tw;
    _Float16* op = y + t * CDIM + cs;
    const f4v s0 = *(const f4v*)&sh[tw * CDIM + cs];
    const f4v s1 = *(const f4v*)&sh[tw * CDIM + cs + 4];
    const f4v s2 = *(const f4v*)&sh[tw * CDIM + cs + 8];
    const f4v s3 = *(const f4v*)&sh[tw * CDIM + cs + 12];
    const half4 h0 = __builtin_convertvector(s0, half4);
    const half4 h1 = __builtin_convertvector(s1, half4);
    const half4 h2 = __builtin_convertvector(s2, half4);
    const half4 h3 = __builtin_convertvector(s3, half4);
    *(half8*)op       = __builtin_shufflevector(h0, h1, 0, 1, 2, 3, 4, 5, 6, 7);
    *(half8*)(op + 8) = __builtin_shufflevector(h2, h3, 0, 1, 2, 3, 4, 5, 6, 7);
}

// ---------------------------------------------------------------------------
// WMMA GEMM: Out[T][N] = epi( A[T][K] @ Bpacked + bias )  (f32 accumulate)
// Block = 128 threads (4 waves), 64x64 output tile; wave = 16 rows x 4 ntiles.
// ---------------------------------------------------------------------------
#define EPI_NONE 0
#define EPI_RELU 1
#define EPI_GELU 2
#define EPI_RES  3

template <int K, bool A_HALF, int EPI, bool OUT_HALF>
__global__ __launch_bounds__(128)
void gemm_wmma(const void* __restrict__ Av, const _Float16* __restrict__ Bp,
               const float* __restrict__ bias, const float* __restrict__ res,
               void* __restrict__ Out, int N) {
    const int lane  = threadIdx.x & 31;
    const int wave  = threadIdx.x >> 5;
    const int mbase = blockIdx.x * 64 + wave * 16;
    const int nbase = blockIdx.y * 64;
    const int mrow  = mbase + (lane & 15);
    const int koff  = (lane >> 4) * 8;     // A-matrix K sub-offset per ISA layout
    const int NT    = N >> 4;
    const int nt0   = nbase >> 4;

    v8f acc[4] = {v8f{}, v8f{}, v8f{}, v8f{}};

    #pragma unroll 4
    for (int kt = 0; kt < K / 32; ++kt) {
        const int k0 = kt * 32;
        v16h afrag;
        if constexpr (A_HALF) {
            const _Float16* A = (const _Float16*)Av + (size_t)mrow * K + k0 + koff;
            const half8 a0 = *(const half8*)(A);
            const half8 a1 = *(const half8*)(A + 16);
            afrag = combine16(a0, a1);
            if (kt + 1 < K / 32) __builtin_prefetch(A + 32, 0, 0);
        } else {
            const float* A = (const float*)Av + (size_t)mrow * K + k0 + koff;
            const f4v f0 = *(const f4v*)(A);
            const f4v f1 = *(const f4v*)(A + 4);
            const f4v f2 = *(const f4v*)(A + 16);
            const f4v f3 = *(const f4v*)(A + 20);
            const v8f fa = __builtin_shufflevector(f0, f1, 0, 1, 2, 3, 4, 5, 6, 7);
            const v8f fb = __builtin_shufflevector(f2, f3, 0, 1, 2, 3, 4, 5, 6, 7);
            afrag = combine16(__builtin_convertvector(fa, half8),
                              __builtin_convertvector(fb, half8));
            if (kt + 1 < K / 32) __builtin_prefetch(A + 32, 0, 0);
        }
        #pragma unroll
        for (int nt = 0; nt < 4; ++nt) {
            const _Float16* bp = Bp + ((size_t)(kt * NT + nt0 + nt) * 32 + lane) * 16;
            const v16h bfrag = *(const v16h*)bp;
            acc[nt] = __builtin_amdgcn_wmma_f32_16x16x32_f16(
                false, afrag, false, bfrag, (short)0, acc[nt], false, false);
        }
    }

    const int row0 = mbase + (lane >> 4) * 8;   // C/D layout: VGPR i -> row0+i
    #pragma unroll
    for (int nt = 0; nt < 4; ++nt) {
        const int col = nbase + nt * 16 + (lane & 15);
        const float bval = bias[col];
        #pragma unroll
        for (int i = 0; i < 8; ++i) {
            const int r = row0 + i;
            float v = acc[nt][i] + bval;
            if constexpr (EPI == EPI_RELU) v = fmaxf(v, 0.0f);
            if constexpr (EPI == EPI_GELU) v = gelu_f(v);
            if constexpr (EPI == EPI_RES) {
                const int b = r / LSEQ, l = r % LSEQ;
                v += res[((size_t)b * CDIM + col) * LSEQ + l];
            }
            if constexpr (OUT_HALF)
                ((_Float16*)Out)[(size_t)r * N + col] = (_Float16)v;
            else
                ((float*)Out)[(size_t)r * N + col] = v;
        }
    }
}

// ---------------------------------------------------------------------------
// kv_partial: per (b,head), split-K WMMA reduction  kv[d][e] = sum_l k[l,d] v[l,e]
// grid = (64, 8 splits), 32 threads (1 wave). A = K^T (gather), B = V (gather).
// ---------------------------------------------------------------------------
__global__ __launch_bounds__(32) void kv_partial(const _Float16* __restrict__ Kh,
                                                 const _Float16* __restrict__ Vh,
                                                 float* __restrict__ kvp) {
    const int lane = threadIdx.x;
    const int bh = blockIdx.x, s = blockIdx.y;
    const int b = bh >> 3, h = bh & 7;
    const size_t base = (size_t)b * LSEQ * CDIM + h * DH;
    const int da = lane & 15, ka = (lane >> 4) * 16;  // A: m=d, 16 K contiguous
    const int eb = lane & 15, kb = (lane >> 4) * 16;  // B: n=e, 16 K contiguous
    v8f acc = v8f{};
    const int l0 = s * (LSEQ / 8);
    for (int ks = 0; ks < (LSEQ / 8) / 32; ++ks) {
        const int lr = l0 + ks * 32;
        v16h a, bf;
        #pragma unroll
        for (int j = 0; j < 8; ++j) {
            a[j]     = Kh[base + (size_t)(lr + ka + j) * CDIM + da];
            a[j + 8] = Kh[base + (size_t)(lr + ka + 8 + j) * CDIM + da];
        }
        #pragma unroll
        for (int j = 0; j < 16; ++j)
            bf[j] = Vh[base + (size_t)(lr + kb + j) * CDIM + eb];
        acc = __builtin_amdgcn_wmma_f32_16x16x32_f16(false, a, false, bf,
                                                     (short)0, acc, false, false);
    }
    float* p = kvp + ((size_t)s * 64 + bh) * 256;
    #pragma unroll
    for (int i = 0; i < 8; ++i)
        p[(i + (lane >> 4) * 8) * 16 + (lane & 15)] = acc[i];
}

// kv_reduce: sum 8 split partials -> kvf[bh][16][16]; also ksum[bh][16].
__global__ __launch_bounds__(256) void kv_reduce(const float* __restrict__ kvp,
                                                 const _Float16* __restrict__ Kh,
                                                 float* __restrict__ kvf,
                                                 float* __restrict__ ksum) {
    const int bh = blockIdx.x, tid = threadIdx.x;
    float s = 0.f;
    #pragma unroll
    for (int sp = 0; sp < 8; ++sp) s += kvp[((size_t)sp * 64 + bh) * 256 + tid];
    kvf[(size_t)bh * 256 + tid] = s;

    __shared__ float part[256];
    const int d = tid & 15, chunk = tid >> 4;
    const int b = bh >> 3, h = bh & 7;
    const _Float16* kp = Kh + (size_t)b * LSEQ * CDIM + h * DH + d;
    float ks = 0.f;
    for (int l = chunk * (LSEQ / 16); l < (chunk + 1) * (LSEQ / 16); ++l)
        ks += (float)kp[(size_t)l * CDIM];
    part[tid] = ks;
    __syncthreads();
    if (tid < 16) {
        float t = 0.f;
        #pragma unroll
        for (int c = 0; c < 16; ++c) t += part[c * 16 + tid];
        ksum[bh * 16 + tid] = t;
    }
}

// attn_apply: out[t][h*16+e] = (sum_d q[d]*kv[d][e]) / (q.ksum + eps)
__global__ __launch_bounds__(256) void attn_apply(const _Float16* __restrict__ Qh,
                                                  const float* __restrict__ kvf,
                                                  const float* __restrict__ ksum,
                                                  _Float16* __restrict__ attn) {
    __shared__ float skv[HEADS * 256];
    __shared__ float sks[HEADS * DH];
    const size_t t = (size_t)blockIdx.x * 256 + threadIdx.x;
    const int bb = (int)(((size_t)blockIdx.x * 256) / LSEQ);  // block-uniform
    for (int i = threadIdx.x; i < HEADS * 256; i += 256)
        skv[i] = kvf[(size_t)bb * HEADS * 256 + i];
    for (int i = threadIdx.x; i < HEADS * DH; i += 256)
        sks[i] = ksum[bb * HEADS * DH + i];
    __syncthreads();

    const _Float16* q = Qh + t * CDIM;
    _Float16* o = attn + t * CDIM;
    #pragma unroll
    for (int h = 0; h < HEADS; ++h) {
        const half8 q0 = *(const half8*)(q + h * DH);
        const half8 q1 = *(const half8*)(q + h * DH + 8);
        const v8f qf0 = __builtin_convertvector(q0, v8f);
        const v8f qf1 = __builtin_convertvector(q1, v8f);
        float qv[DH], den = 0.f;
        #pragma unroll
        for (int d = 0; d < 8; ++d) {
            qv[d] = qf0[d]; qv[d + 8] = qf1[d];
        }
        #pragma unroll
        for (int d = 0; d < DH; ++d) den += qv[d] * sks[h * DH + d];
        const float inv = 1.0f / (den + 1e-6f);
        v8f nm0 = v8f{}, nm1 = v8f{};
        #pragma unroll
        for (int d = 0; d < DH; ++d) {
            const float qd = qv[d];
            const f4v k0 = *(const f4v*)&skv[h * 256 + d * 16];
            const f4v k1 = *(const f4v*)&skv[h * 256 + d * 16 + 4];
            const f4v k2 = *(const f4v*)&skv[h * 256 + d * 16 + 8];
            const f4v k3 = *(const f4v*)&skv[h * 256 + d * 16 + 12];
            const v8f ka = __builtin_shufflevector(k0, k1, 0, 1, 2, 3, 4, 5, 6, 7);
            const v8f kb = __builtin_shufflevector(k2, k3, 0, 1, 2, 3, 4, 5, 6, 7);
            nm0 += qd * ka;
            nm1 += qd * kb;
        }
        *(half8*)(o + h * DH)     = __builtin_convertvector(nm0 * inv, half8);
        *(half8*)(o + h * DH + 8) = __builtin_convertvector(nm1 * inv, half8);
    }
}

// ---------------------------------------------------------------------------
// Residual add + LayerNorm (warp per token). NCHW_OUT writes final d_out.
// ---------------------------------------------------------------------------
template <bool NCHW_OUT>
__global__ __launch_bounds__(256) void add_ln(const float* __restrict__ A,
                                              const float* __restrict__ Bt,
                                              const float* __restrict__ g,
                                              const float* __restrict__ be,
                                              float* __restrict__ out) {
    const int warp = threadIdx.x >> 5, lane = threadIdx.x & 31;
    const size_t t = (size_t)blockIdx.x * 8 + warp;
    const f4v va = ((const f4v*)(A + t * CDIM))[lane];
    const f4v vb = ((const f4v*)(Bt + t * CDIM))[lane];
    float v[4], sum = 0.f;
    #pragma unroll
    for (int j = 0; j < 4; ++j) { v[j] = va[j] + vb[j]; sum += v[j]; }
    #pragma unroll
    for (int m = 16; m >= 1; m >>= 1) sum += __shfl_xor(sum, m, 32);
    const float mean = sum * (1.0f / CDIM);
    float vs = 0.f;
    #pragma unroll
    for (int j = 0; j < 4; ++j) { float d = v[j] - mean; vs += d * d; }
    #pragma unroll
    for (int m = 16; m >= 1; m >>= 1) vs += __shfl_xor(vs, m, 32);
    const float rstd = rsqrtf(vs * (1.0f / CDIM) + 1e-5f);
    #pragma unroll
    for (int j = 0; j < 4; ++j) {
        const int c = lane * 4 + j;
        const float o = (v[j] - mean) * rstd * g[c] + be[c];
        if constexpr (NCHW_OUT) {
            const int b = (int)(t / LSEQ), l = (int)(t % LSEQ);
            out[((size_t)b * CDIM + c) * LSEQ + l] = o;
        } else {
            out[t * CDIM + c] = o;
        }
    }
}

// ---------------------------------------------------------------------------
extern "C" void kernel_launch(void* const* d_in, const int* in_sizes, int n_in,
                              void* d_out, int out_size, void* d_ws, size_t ws_size,
                              hipStream_t stream) {
    const float* ref  = (const float*)d_in[0];
    const float* oth  = (const float*)d_in[1];
    const float* dw_w = (const float*)d_in[2];
    const float* dw_b = (const float*)d_in[3];
    const float* pw_w = (const float*)d_in[4];
    const float* pw_b = (const float*)d_in[5];
    const float* wq = (const float*)d_in[6],  *bq = (const float*)d_in[7];
    const float* wk = (const float*)d_in[8],  *bk = (const float*)d_in[9];
    const float* wv = (const float*)d_in[10], *bv = (const float*)d_in[11];
    const float* wo = (const float*)d_in[12], *bo = (const float*)d_in[13];
    const float* w1 = (const float*)d_in[14], *b1 = (const float*)d_in[15];
    const float* w2 = (const float*)d_in[16], *b2 = (const float*)d_in[17];
    const float* ln1g = (const float*)d_in[18], *ln1b = (const float*)d_in[19];
    const float* ln2g = (const float*)d_in[20], *ln2b = (const float*)d_in[21];

    char* ws = (char*)d_ws;
    size_t off = 0;
    auto alloc = [&](size_t n) { size_t o = off; off = (off + n + 255) & ~(size_t)255; return o; };

    const size_t HBUF = (size_t)T_TOK * CDIM * sizeof(_Float16);  // 18.87 MB
    const size_t FBUF = (size_t)T_TOK * CDIM * sizeof(float);     // 37.75 MB

    _Float16* Wpw = (_Float16*)(ws + alloc(128 * 128 * 2));
    _Float16* Wq  = (_Float16*)(ws + alloc(128 * 128 * 2));
    _Float16* Wk  = (_Float16*)(ws + alloc(128 * 128 * 2));
    _Float16* Wv  = (_Float16*)(ws + alloc(128 * 128 * 2));
    _Float16* Wo  = (_Float16*)(ws + alloc(128 * 128 * 2));
    _Float16* W1  = (_Float16*)(ws + alloc(128 * 512 * 2));
    _Float16* W2  = (_Float16*)(ws + alloc(512 * 128 * 2));
    float* kvp  = (float*)(ws + alloc(8 * 64 * 256 * 4));
    float* kvf  = (float*)(ws + alloc(64 * 256 * 4));
    float* ksum = (float*)(ws + alloc(64 * 16 * 4));
    _Float16* Y1r  = (_Float16*)(ws + alloc(HBUF));   // later reused as attn
    _Float16* Y1o  = (_Float16*)(ws + alloc(HBUF));   // hid aliases Y1o..Kh (4*HBUF)
    _Float16* othv = (_Float16*)(ws + alloc(HBUF));
    _Float16* Qh   = (_Float16*)(ws + alloc(HBUF));
    _Float16* Kh   = (_Float16*)(ws + alloc(HBUF));
    _Float16* Vh   = (_Float16*)(ws + alloc(HBUF));
    float* refv = (float*)(ws + alloc(FBUF));         // later reused as ffn2
    float* out2 = (float*)(ws + alloc(FBUF));
    float* x1   = (float*)(ws + alloc(FBUF));
    _Float16* attn = Y1r;
    _Float16* hid  = Y1o;                              // [T][512] f16
    float* ffn2    = refv;

    // 1) repack weights into WMMA B-fragment layout
    repack_w<<<dim3(128 * 128 / 256), 256, 0, stream>>>(pw_w, Wpw, 128, 128, 1);
    repack_w<<<dim3(128 * 128 / 256), 256, 0, stream>>>(wq, Wq, 128, 128, 0);
    repack_w<<<dim3(128 * 128 / 256), 256, 0, stream>>>(wk, Wk, 128, 128, 0);
    repack_w<<<dim3(128 * 128 / 256), 256, 0, stream>>>(wv, Wv, 128, 128, 0);
    repack_w<<<dim3(128 * 128 / 256), 256, 0, stream>>>(wo, Wo, 128, 128, 0);
    repack_w<<<dim3(128 * 512 / 256), 256, 0, stream>>>(w1, W1, 128, 512, 0);
    repack_w<<<dim3(512 * 128 / 256), 256, 0, stream>>>(w2, W2, 512, 128, 0);

    // 2) depthwise conv + GELU -> token-major f16
    dwconv_gelu<<<dim3(8 * 96 * 3), 256, 0, stream>>>(ref, dw_w, dw_b, Y1r);
    dwconv_gelu<<<dim3(8 * 96 * 3), 256, 0, stream>>>(oth, dw_w, dw_b, Y1o);

    const dim3 blk(128);
    const dim3 g128(T_TOK / 64, 128 / 64);
    const dim3 g512(T_TOK / 64, 512 / 64);

    // 3) pointwise conv + bias + NCHW residual (EfficientViT block output)
    gemm_wmma<128, true, EPI_RES, false><<<g128, blk, 0, stream>>>(Y1r, Wpw, pw_b, ref, refv, 128);
    gemm_wmma<128, true, EPI_RES, true ><<<g128, blk, 0, stream>>>(Y1o, Wpw, pw_b, oth, othv, 128);

    // 4) Q/K/V projections (ReLU fused for q, k)
    gemm_wmma<128, false, EPI_RELU, true><<<g128, blk, 0, stream>>>(refv, Wq, bq, nullptr, Qh, 128);
    gemm_wmma<128, true,  EPI_RELU, true><<<g128, blk, 0, stream>>>(othv, Wk, bk, nullptr, Kh, 128);
    gemm_wmma<128, true,  EPI_NONE, true><<<g128, blk, 0, stream>>>(othv, Wv, bv, nullptr, Vh, 128);

    // 5) linear attention: kv = k^T v (split-K WMMA), ksum, then apply
    kv_partial<<<dim3(64, 8), 32, 0, stream>>>(Kh, Vh, kvp);
    kv_reduce<<<dim3(64), 256, 0, stream>>>(kvp, Kh, kvf, ksum);
    attn_apply<<<dim3(T_TOK / 256), 256, 0, stream>>>(Qh, kvf, ksum, attn);

    // 6) output projection, residual + LN1
    gemm_wmma<128, true, EPI_NONE, false><<<g128, blk, 0, stream>>>(attn, Wo, bo, nullptr, out2, 128);
    add_ln<false><<<dim3(T_TOK / 8), 256, 0, stream>>>(refv, out2, ln1g, ln1b, x1);

    // 7) FFN: 128 -> 512 (GELU) -> 128, residual + LN2 -> NCHW d_out
    gemm_wmma<128, false, EPI_GELU, true ><<<g512, blk, 0, stream>>>(x1, W1, b1, nullptr, hid, 512);
    gemm_wmma<512, true,  EPI_NONE, false><<<g128, blk, 0, stream>>>(hid, W2, b2, nullptr, ffn2, 128);
    add_ln<true><<<dim3(T_TOK / 8), 256, 0, stream>>>(x1, ffn2, ln2g, ln2b, (float*)d_out);
}
